// QDGCNN_cls_61890478735427
// MI455X (gfx1250) — matrix operand order repaired and checked by reading, not compiled, not run on hardware
//
#include <hip/hip_runtime.h>

// ---------------- problem constants (from reference) ----------------
#define BB   8
#define NN   1024
#define KNB  20
#define NKC  (NN * KNB)   // 20480 edge columns per batch item
#define EPSV 1e-5f

typedef __attribute__((ext_vector_type(16))) __bf16 v16bf;
typedef __attribute__((ext_vector_type(8)))  float  v8f;
typedef __attribute__((ext_vector_type(4)))  unsigned int u32x4;
typedef __attribute__((ext_vector_type(8)))  unsigned int u32x8;

// ---------------- helpers ----------------
// sign-aware float atomic max (works with -inf init)
__device__ inline void atomicMaxF(float* addr, float v) {
  if (v >= 0.0f) atomicMax((int*)addr, __float_as_int(v));
  else           atomicMin((unsigned int*)addr, __float_as_uint(v));
}

// Tensor Data Mover: 1-D DMA of `ndwords` dwords from global `src` into LDS
// at byte offset `lds_off`. Descriptor per CDNA5 ISA
// (group0: count/lds_addr/global_addr/type=2; group1: data_size=4B, 1-D tile).
__device__ inline void tdm_load_lds(unsigned lds_off, const void* src, unsigned ndwords) {
  unsigned long long ga = (unsigned long long)(size_t)src;
  u32x4 g0;
  g0[0] = 1u;                                                  // count=1, user mode
  g0[1] = lds_off;                                             // lds_addr
  g0[2] = (unsigned)ga;                                        // global_addr[31:0]
  g0[3] = (unsigned)((ga >> 32) & 0x01FFFFFFull) | (2u << 30); // addr[56:32] | type=2
  u32x8 g1;
  g1[0] = 0x00020000u;                   // workgroup_mask=0, data_size=2 (4 bytes)
  g1[1] = (ndwords & 0xFFFFu) << 16;     // tensor_dim0[15:0]
  g1[2] = (ndwords >> 16) | (1u << 16);  // tensor_dim0[31:16] | tensor_dim1=1
  g1[3] = (ndwords & 0xFFFFu) << 16;     // tile_dim0 = ndwords (<=65535)
  g1[4] = 0u;                            // tile_dim1/2 unused
  g1[5] = ndwords;                       // tensor_dim0_stride[31:0]
  g1[6] = 0u;
  g1[7] = 0u;
  asm volatile("tensor_load_to_lds %0, %1" :: "s"(g0), "s"(g1) : "memory");
}
__device__ inline void tdm_wait() {
  asm volatile("s_wait_tensorcnt 0x0" ::: "memory");
}

// ---------------- weight pre-pack into WMMA A-fragment order (bf16) -------
// Entry (mt, ks, lane) holds the 16 bf16 values lane `lane` needs for
// m-tile `mt`, K-step `ks`:  v[i] -> K = ks*32 + 16*hi + i.
__global__ void pack_a_kernel(const float* __restrict__ W, int C2, int cout,
                              __bf16* __restrict__ P, int ksteps) {
  int t = blockIdx.x * blockDim.x + threadIdx.x;
  int total = (cout / 16) * ksteps * 32;
  if (t >= total) return;
  int lane = t & 31;
  int ks = (t >> 5) % ksteps;
  int mt = t / (32 * ksteps);
  int lo = lane & 15, hi = lane >> 4;
  int m = mt * 16 + lo, kk = ks * 32;
  v16bf a;
#pragma unroll
  for (int i = 0; i < 16; ++i) {
    int k = kk + 16 * hi + i;
    float v = (k < C2) ? W[(size_t)m * C2 + k] : 0.0f;
    a[i] = (__bf16)v;
  }
  *(v16bf*)(P + (size_t)t * 16) = a;
}

// ---------------- feature transpose (b,c,n) f32 -> (b,n,cP) bf16 ----------
template <int CIN>
__global__ void xpose_kernel(const float* __restrict__ in, int in_bstride,
                             __bf16* __restrict__ xt) {
  constexpr int CP = (CIN + 15) & ~15;
  int t = blockIdx.x * blockDim.x + threadIdx.x;
  if (t >= BB * NN * CP) return;
  int c = t % CP;
  int bn = t / CP;
  int b = bn >> 10, n = bn & (NN - 1);
  float v = (c < CIN) ? in[(size_t)b * in_bstride + (size_t)c * NN + n] : 0.0f;
  xt[t] = (__bf16)v;
}

__global__ void hxpose_kernel(const float* __restrict__ hcat, __bf16* __restrict__ ht) {
  int t = blockIdx.x * blockDim.x + threadIdx.x;
  if (t >= BB * NN * 512) return;
  int c = t & 511;
  int bn = t >> 9;
  int b = bn >> 10, n = bn & (NN - 1);
  ht[t] = (__bf16)hcat[((size_t)b * 512 + c) * NN + n];
}

// ---------------- edge-conv GEMM (r,g,q fused), WMMA bf16 + TDM ----------
// MODE 0: accumulate BN stats.  MODE 1: BN + leaky + atomic-max over k.
template <int CIN, int MODE>
__global__ __launch_bounds__(32)
void edge_gemm(const __bf16* __restrict__ xt,
               const int* __restrict__ idx,
               const __bf16* __restrict__ pR, const __bf16* __restrict__ pG,
               const __bf16* __restrict__ pB,
               const float* __restrict__ bgv, const float* __restrict__ gam,
               const float* __restrict__ bet,
               float* __restrict__ stats, int cout,
               float* __restrict__ outp, size_t out_bstride) {
  constexpr int C2 = 2 * CIN;
  constexpr int KSTEPS = (C2 + 31) / 32;
  constexpr int CP = (CIN + 15) & ~15;
  constexpr unsigned FRAG_DW = KSTEPS * 32 * 8;   // dwords per matrix per m-tile
  extern __shared__ char smem_raw[];

  const int lane = threadIdx.x;
  const int lo = lane & 15, hi = lane >> 4;
  const int b  = blockIdx.z;
  const int mt = blockIdx.y;
  const int m0 = mt * 16;
  const int col = blockIdx.x * 16 + lo;
  const int n  = col / KNB;
  const int kq = col - n * KNB;

  // DMA this m-tile's packed A-fragments (3 matrices) into LDS
  const size_t fragElems = (size_t)KSTEPS * 32 * 16;
  tdm_load_lds(0u,           pR + (size_t)mt * fragElems, FRAG_DW);
  tdm_load_lds(FRAG_DW * 4,  pG + (size_t)mt * fragElems, FRAG_DW);
  tdm_load_lds(FRAG_DW * 8,  pB + (size_t)mt * fragElems, FRAG_DW);

  const int nbr = idx[((size_t)b * NN + n) * KNB + kq];
  const __bf16* cen = xt + (size_t)(b * NN + n) * CP;
  const __bf16* nb  = xt + (size_t)(b * NN + nbr) * CP;

  tdm_wait();
  const v16bf* fragR = (const v16bf*)(smem_raw);
  const v16bf* fragG = fragR + KSTEPS * 32;
  const v16bf* fragB = fragG + KSTEPS * 32;

  v8f accR = {}, accG = {}, accQ = {};
#pragma unroll
  for (int ks = 0; ks < KSTEPS; ++ks) {
    v16bf aR = fragR[ks * 32 + lane];
    v16bf aG = fragG[ks * 32 + lane];
    v16bf aB = fragB[ks * 32 + lane];
    v16bf be, be2;
    const int c0 = ks * 32 + 16 * hi;
    if constexpr (CIN % 16 == 0) {
      if (c0 < CIN) {            // diff region (never straddles: 16 | CIN)
        v16bf xn = *(const v16bf*)(nb + c0);
        v16bf xc = *(const v16bf*)(cen + c0);
#pragma unroll
        for (int i = 0; i < 16; ++i) {
          float d = (float)xn[i] - (float)xc[i];
          be[i] = (__bf16)d;
          be2[i] = (__bf16)(d * d);
        }
      } else {                   // center-broadcast region
        v16bf xc = *(const v16bf*)(cen + (c0 - CIN));
#pragma unroll
        for (int i = 0; i < 16; ++i) {
          float v = (float)xc[i];
          be[i] = xc[i];
          be2[i] = (__bf16)(v * v);
        }
      }
    } else {                     // tiny C2 (block 1): generic, compile-time folds
#pragma unroll
      for (int i = 0; i < 16; ++i) {
        int c = c0 + i;
        float v = 0.0f;
        if (c < CIN) v = (float)nb[c] - (float)cen[c];
        else if (c < C2) v = (float)cen[c - CIN];
        be[i] = (__bf16)v;
        be2[i] = (__bf16)(v * v);
      }
    }
    accR = __builtin_amdgcn_wmma_f32_16x16x32_bf16(false, aR, false, be,  (short)0, accR, false, false);
    accG = __builtin_amdgcn_wmma_f32_16x16x32_bf16(false, aG, false, be,  (short)0, accG, false, false);
    accQ = __builtin_amdgcn_wmma_f32_16x16x32_bf16(false, aB, false, be2, (short)0, accQ, false, false);
  }

  if (MODE == 0) {
#pragma unroll
    for (int v = 0; v < 8; ++v) {
      int M = m0 + v + 8 * hi;
      float z = accR[v] * (accG[v] + bgv[M]) + accQ[v];
      float s = z, q = z * z;
      for (int off = 1; off < 16; off <<= 1) {
        s += __shfl_xor(s, off, 32);
        q += __shfl_xor(q, off, 32);
      }
      if (lo == 0) {
        atomicAdd(&stats[M], s);
        atomicAdd(&stats[cout + M], q);
      }
    }
  } else {
#pragma unroll
    for (int v = 0; v < 8; ++v) {
      int M = m0 + v + 8 * hi;
      float z = accR[v] * (accG[v] + bgv[M]) + accQ[v];
      float h = (z - stats[M]) * stats[cout + M] * gam[M] + bet[M];
      h = (h >= 0.0f) ? h : 0.2f * h;
      atomicMaxF(&outp[(size_t)b * out_bstride + (size_t)M * NN + n], h);
    }
  }
}

// ---------------- embedding conv GEMM (1024 x 512), WMMA bf16 + TDM ------
template <int MODE>
__global__ __launch_bounds__(32)
void emb_gemm(const __bf16* __restrict__ ht,
              const __bf16* __restrict__ pR, const __bf16* __restrict__ pG,
              const __bf16* __restrict__ pB,
              const float* __restrict__ bgv, const float* __restrict__ gam,
              const float* __restrict__ bet,
              float* __restrict__ stats, float* __restrict__ emb) {
  constexpr int KSTEPS = 16;            // C2 = 512
  constexpr unsigned FRAG_DW = KSTEPS * 32 * 8;
  extern __shared__ char smem_raw[];

  const int lane = threadIdx.x;
  const int lo = lane & 15, hi = lane >> 4;
  const int mt = blockIdx.y;
  const int m0 = mt * 16;
  const int col = blockIdx.x * 16 + lo;  // b*N + n
  const int b = col >> 10, n = col & (NN - 1);

  const size_t fragElems = (size_t)KSTEPS * 32 * 16;
  tdm_load_lds(0u,          pR + (size_t)mt * fragElems, FRAG_DW);
  tdm_load_lds(FRAG_DW * 4, pG + (size_t)mt * fragElems, FRAG_DW);
  tdm_load_lds(FRAG_DW * 8, pB + (size_t)mt * fragElems, FRAG_DW);

  const __bf16* hp = ht + (size_t)col * 512;

  tdm_wait();
  const v16bf* fragR = (const v16bf*)(smem_raw);
  const v16bf* fragG = fragR + KSTEPS * 32;
  const v16bf* fragB = fragG + KSTEPS * 32;

  v8f accR = {}, accG = {}, accQ = {};
#pragma unroll
  for (int ks = 0; ks < KSTEPS; ++ks) {
    v16bf aR = fragR[ks * 32 + lane];
    v16bf aG = fragG[ks * 32 + lane];
    v16bf aB = fragB[ks * 32 + lane];
    const int c0 = ks * 32 + 16 * hi;
    v16bf hv = *(const v16bf*)(hp + c0);
    v16bf be2;
#pragma unroll
    for (int i = 0; i < 16; ++i) {
      float v = (float)hv[i];
      be2[i] = (__bf16)(v * v);
    }
    accR = __builtin_amdgcn_wmma_f32_16x16x32_bf16(false, aR, false, hv,  (short)0, accR, false, false);
    accG = __builtin_amdgcn_wmma_f32_16x16x32_bf16(false, aG, false, hv,  (short)0, accG, false, false);
    accQ = __builtin_amdgcn_wmma_f32_16x16x32_bf16(false, aB, false, be2, (short)0, accQ, false, false);
  }

  if (MODE == 0) {
#pragma unroll
    for (int v = 0; v < 8; ++v) {
      int M = m0 + v + 8 * hi;
      float z = accR[v] * (accG[v] + bgv[M]) + accQ[v];
      float s = z, q = z * z;
      for (int off = 1; off < 16; off <<= 1) {
        s += __shfl_xor(s, off, 32);
        q += __shfl_xor(q, off, 32);
      }
      if (lo == 0) {
        atomicAdd(&stats[M], s);
        atomicAdd(&stats[1024 + M], q);
      }
    }
  } else {
#pragma unroll
    for (int v = 0; v < 8; ++v) {
      int M = m0 + v + 8 * hi;
      float z = accR[v] * (accG[v] + bgv[M]) + accQ[v];
      float h = (z - stats[M]) * stats[1024 + M] * gam[M] + bet[M];
      h = (h >= 0.0f) ? h : 0.2f * h;
      emb[((size_t)b * 1024 + M) * NN + n] = h;
    }
  }
}

// ---------------- KNN ----------------
template <int CIN>
__global__ void xx_kernel(const float* __restrict__ in, int in_bstride,
                          float* __restrict__ xx) {
  int t = blockIdx.x * blockDim.x + threadIdx.x;
  if (t >= BB * NN) return;
  int b = t >> 10, n = t & (NN - 1);
  const float* xb = in + (size_t)b * in_bstride;
  float s = 0.0f;
#pragma unroll 4
  for (int c = 0; c < CIN; ++c) { float v = xb[(size_t)c * NN + n]; s += v * v; }
  xx[t] = s;
}

template <int CIN>
__global__ void knn_kernel(const float* __restrict__ in, int in_bstride,
                           const float* __restrict__ xx, int* __restrict__ idx) {
  int t = blockIdx.x * blockDim.x + threadIdx.x;
  if (t >= BB * NN) return;
  int b = t >> 10, n = t & (NN - 1);
  const float* xb = in + (size_t)b * in_bstride;
  float xn[CIN];
#pragma unroll 4
  for (int c = 0; c < CIN; ++c) xn[c] = xb[(size_t)c * NN + n];
  const float xxn = xx[t];
  const float* xxb = xx + (size_t)b * NN;

  float bestd[KNB]; int besti[KNB];
  for (int j = 0; j < KNB; ++j) { bestd[j] = -3.4e38f; besti[j] = 0; }
  for (int m = 0; m < NN; ++m) {
    float inner = 0.0f;
#pragma unroll 4
    for (int c = 0; c < CIN; ++c) inner += xn[c] * xb[(size_t)c * NN + m];
    float d = 2.0f * inner - xxn - xxb[m];   // -||xi-xj||^2
    if (d > bestd[KNB - 1]) {
      int p = KNB - 1;
      while (p > 0 && bestd[p - 1] < d) {
        bestd[p] = bestd[p - 1]; besti[p] = besti[p - 1]; --p;
      }
      bestd[p] = d; besti[p] = m;
    }
  }
  for (int j = 0; j < KNB; ++j) idx[(size_t)t * KNB + j] = besti[j];
}

// ---------------- misc small kernels ----------------
__global__ void fill_kernel(float* __restrict__ p, float v, int nelem) {
  int t = blockIdx.x * blockDim.x + threadIdx.x;
  if (t < nelem) p[t] = v;
}

__global__ void finalize_stats(float* __restrict__ stats, int C, float invcnt) {
  int o = blockIdx.x * blockDim.x + threadIdx.x;
  if (o >= C) return;
  float mu  = stats[o] * invcnt;
  float var = stats[C + o] * invcnt - mu * mu;
  stats[o]     = mu;
  stats[C + o] = rsqrtf(var + EPSV);
}

__global__ void pool_kernel(const float* __restrict__ emb, float* __restrict__ pooled) {
  int t = blockIdx.x * blockDim.x + threadIdx.x;
  if (t >= BB * 1024) return;
  int b = t >> 10, o = t & 1023;
  const float* p = emb + (size_t)t * NN;
  float mx = -3.4e38f, s = 0.0f;
  for (int n = 0; n < NN; ++n) { float v = p[n]; mx = fmaxf(mx, v); s += v; }
  pooled[(size_t)b * 2048 + o]        = mx;
  pooled[(size_t)b * 2048 + 1024 + o] = s * (1.0f / NN);
}

__global__ void fc_kernel(const float* __restrict__ inp, int infeat,
                          const float* __restrict__ W, const float* __restrict__ bias,
                          float* __restrict__ outp, int outfeat) {
  int t = blockIdx.x * blockDim.x + threadIdx.x;
  if (t >= BB * outfeat) return;
  int b = t / outfeat, o = t - b * outfeat;
  float s = bias ? bias[o] : 0.0f;
  const float* wi = W + (size_t)o * infeat;
  const float* xi = inp + (size_t)b * infeat;
  for (int i = 0; i < infeat; ++i) s += xi[i] * wi[i];
  outp[t] = s;
}

__global__ void bnact_kernel(float* __restrict__ y, int F,
                             const float* __restrict__ gam, const float* __restrict__ bet) {
  int o = blockIdx.x * blockDim.x + threadIdx.x;
  if (o >= F) return;
  float s = 0.0f, q = 0.0f;
  for (int b = 0; b < BB; ++b) { float v = y[(size_t)b * F + o]; s += v; q += v * v; }
  float mu = s * (1.0f / BB);
  float rs = rsqrtf(q * (1.0f / BB) - mu * mu + EPSV);
  for (int b = 0; b < BB; ++b) {
    float v = y[(size_t)b * F + o];
    float h = (v - mu) * rs * gam[o] + bet[o];
    y[(size_t)b * F + o] = (h >= 0.0f) ? h : 0.2f * h;
  }
}

// ---------------- host-side per-block driver ----------------
template <int CIN>
static void run_edge_block(const float* in, int in_bstride, int cout,
                           const float* wr, const float* wg, const float* bg,
                           const float* wb, const float* gm, const float* bt,
                           __bf16* pR, __bf16* pG, __bf16* pB, __bf16* xt,
                           int* idxb, float* xxb, float* stats,
                           float* hcat, int coff, hipStream_t stream) {
  constexpr int C2 = 2 * CIN;
  constexpr int KS = (C2 + 31) / 32;
  constexpr int CP = (CIN + 15) & ~15;

  {
    int total = BB * NN * CP;
    xpose_kernel<CIN><<<(total + 255) / 256, 256, 0, stream>>>(in, in_bstride, xt);
  }
  xx_kernel<CIN><<<(BB * NN + 255) / 256, 256, 0, stream>>>(in, in_bstride, xxb);
  knn_kernel<CIN><<<(BB * NN + 127) / 128, 128, 0, stream>>>(in, in_bstride, xxb, idxb);

  int ptotal = (cout / 16) * KS * 32;
  pack_a_kernel<<<(ptotal + 255) / 256, 256, 0, stream>>>(wr, C2, cout, pR, KS);
  pack_a_kernel<<<(ptotal + 255) / 256, 256, 0, stream>>>(wg, C2, cout, pG, KS);
  pack_a_kernel<<<(ptotal + 255) / 256, 256, 0, stream>>>(wb, C2, cout, pB, KS);

  fill_kernel<<<(2 * cout + 255) / 256, 256, 0, stream>>>(stats, 0.0f, 2 * cout);

  dim3 g(NKC / 16, cout / 16, BB);
  size_t shm = (size_t)3 * KS * 1024;   // 3 matrices * KS*32 entries * 32 B
  edge_gemm<CIN, 0><<<g, 32, shm, stream>>>(xt, idxb, pR, pG, pB, bg, gm, bt,
                                            stats, cout, nullptr, 0);
  finalize_stats<<<(cout + 255) / 256, 256, 0, stream>>>(
      stats, cout, 1.0f / ((float)BB * NN * KNB));
  edge_gemm<CIN, 1><<<g, 32, shm, stream>>>(xt, idxb, pR, pG, pB, bg, gm, bt,
                                            stats, cout,
                                            hcat + (size_t)coff * NN, (size_t)512 * NN);
}

// ---------------- launcher ----------------
extern "C" void kernel_launch(void* const* d_in, const int* in_sizes, int n_in,
                              void* d_out, int out_size, void* d_ws, size_t ws_size,
                              hipStream_t stream) {
  (void)in_sizes; (void)n_in; (void)out_size; (void)ws_size;
  const float* x = (const float*)d_in[0];

  char* ws = (char*)d_ws;
  size_t off = 0;
  auto alloc = [&](size_t bytes) -> void* {
    void* p = ws + off;
    off += (bytes + 255) & ~(size_t)255;
    return p;
  };
  int*    idxb   = (int*)alloc((size_t)BB * NN * KNB * 4);
  float*  hcat   = (float*)alloc((size_t)BB * 512 * NN * 4);
  float*  emb    = (float*)alloc((size_t)BB * 1024 * NN * 4);
  float*  stats  = (float*)alloc((size_t)2 * 1024 * 4);
  float*  xxb    = (float*)alloc((size_t)BB * NN * 4);
  float*  pooled = (float*)alloc((size_t)BB * 2048 * 4);
  float*  y1     = (float*)alloc((size_t)BB * 512 * 4);
  float*  y2     = (float*)alloc((size_t)BB * 256 * 4);
  __bf16* xt     = (__bf16*)alloc((size_t)BB * NN * 128 * 2);   // max CP=128
  __bf16* ht     = (__bf16*)alloc((size_t)BB * NN * 512 * 2);
  // packed A-fragments: (cout/16) * ksteps * 32 entries * 16 bf16, x3 matrices
  auto allocPk = [&](int cout, int C2) -> __bf16* {
    int ks = (C2 + 31) / 32;
    return (__bf16*)alloc((size_t)(cout / 16) * ks * 32 * 16 * 2);
  };
  __bf16 *p1R = allocPk(64, 32),    *p1G = allocPk(64, 32),    *p1B = allocPk(64, 32);
  __bf16 *p2R = allocPk(64, 128),   *p2G = allocPk(64, 128),   *p2B = allocPk(64, 128);
  __bf16 *p3R = allocPk(128, 128),  *p3G = allocPk(128, 128),  *p3B = allocPk(128, 128);
  __bf16 *p4R = allocPk(256, 256),  *p4G = allocPk(256, 256),  *p4B = allocPk(256, 256);
  __bf16 *p5R = allocPk(1024, 512), *p5G = allocPk(1024, 512), *p5B = allocPk(1024, 512);

  // init concat buffer to -inf (every element receives k atomic-max contributions)
  {
    int nelem = BB * 512 * NN;
    fill_kernel<<<(nelem + 255) / 256, 256, 0, stream>>>(hcat, -3.4e38f, nelem);
  }

  const float** P = (const float**)d_in;   // convenience view

  run_edge_block<3>(x, 3 * NN, 64, P[1], P[2], P[3], P[4], P[5], P[6],
                    p1R, p1G, p1B, xt, idxb, xxb, stats, hcat, 0, stream);
  run_edge_block<64>(hcat, 512 * NN, 64, P[7], P[8], P[9], P[10], P[11], P[12],
                     p2R, p2G, p2B, xt, idxb, xxb, stats, hcat, 64, stream);
  run_edge_block<64>(hcat + 64 * NN, 512 * NN, 128, P[13], P[14], P[15], P[16], P[17], P[18],
                     p3R, p3G, p3B, xt, idxb, xxb, stats, hcat, 128, stream);
  run_edge_block<128>(hcat + 128 * NN, 512 * NN, 256, P[19], P[20], P[21], P[22], P[23], P[24],
                      p4R, p4G, p4B, xt, idxb, xxb, stats, hcat, 256, stream);

  // embedding conv
  {
    int total = BB * NN * 512;
    hxpose_kernel<<<(total + 255) / 256, 256, 0, stream>>>(hcat, ht);
    int ptotal = (1024 / 16) * 16 * 32;
    pack_a_kernel<<<(ptotal + 255) / 256, 256, 0, stream>>>(P[25], 512, 1024, p5R, 16);
    pack_a_kernel<<<(ptotal + 255) / 256, 256, 0, stream>>>(P[26], 512, 1024, p5G, 16);
    pack_a_kernel<<<(ptotal + 255) / 256, 256, 0, stream>>>(P[28], 512, 1024, p5B, 16);
    fill_kernel<<<(2 * 1024 + 255) / 256, 256, 0, stream>>>(stats, 0.0f, 2 * 1024);
    dim3 ge(BB * NN / 16, 1024 / 16, 1);
    size_t shm = (size_t)3 * 16 * 1024;   // 48 KB
    emb_gemm<0><<<ge, 32, shm, stream>>>(ht, p5R, p5G, p5B, P[27], P[29], P[30], stats, nullptr);
    finalize_stats<<<(1024 + 255) / 256, 256, 0, stream>>>(stats, 1024, 1.0f / (BB * NN));
    emb_gemm<1><<<ge, 32, shm, stream>>>(ht, p5R, p5G, p5B, P[27], P[29], P[30], stats, emb);
  }

  // pool (max | mean) -> (8, 2048)
  pool_kernel<<<(BB * 1024 + 255) / 256, 256, 0, stream>>>(emb, pooled);

  // FC head
  fc_kernel<<<(BB * 512 + 255) / 256, 256, 0, stream>>>(pooled, 2048, P[31], nullptr, y1, 512);
  bnact_kernel<<<(512 + 255) / 256, 256, 0, stream>>>(y1, 512, P[32], P[33]);
  fc_kernel<<<(BB * 256 + 255) / 256, 256, 0, stream>>>(y1, 512, P[34], P[35], y2, 256);
  bnact_kernel<<<(256 + 255) / 256, 256, 0, stream>>>(y2, 256, P[36], P[37]);
  fc_kernel<<<(BB * 40 + 255) / 256, 256, 0, stream>>>(y2, 256, P[38], P[39],
                                                       (float*)d_out, 40);
}
